// Net1_39118562132400
// MI455X (gfx1250) — matrix operand order repaired and checked by reading, not compiled, not run on hardware
//
#include <hip/hip_runtime.h>
#include <hip/hip_bf16.h>

// ---------------------------------------------------------------------------
// GAT (2-layer, 4 heads x 64) + BN/ReLU + FC for gfx1250 (CDNA5, wave32).
// Dense GEMMs run on v_wmma_f32_16x16x32_bf16 (bf16 in, f32 accumulate).
// ---------------------------------------------------------------------------

typedef __bf16 bf16_t;
typedef __attribute__((ext_vector_type(16))) __bf16 v16bf;
typedef __attribute__((ext_vector_type(8)))  __bf16 v8bf;
typedef __attribute__((ext_vector_type(8)))  float  v8f;

#define HC   256
#define HEADS 4
#define HID  64
#define BIGC 768   // [xs | xd | skip] fused projection width
#define NCLS 349

__device__ __forceinline__ bf16_t f2bf(float f) {
    unsigned u = __float_as_uint(f);
    u = u + 0x7FFFu + ((u >> 16) & 1u);      // round-to-nearest-even
    unsigned short s = (unsigned short)(u >> 16);
    bf16_t r;
    __builtin_memcpy(&r, &s, 2);
    return r;
}

__device__ __forceinline__ void atomicMaxF(float* addr, float v) {
    // sign-split trick: positives monotonic as int, negatives reversed as uint
    if (v >= 0.f) atomicMax((int*)addr, __float_as_int(v));
    else          atomicMin((unsigned int*)addr, __float_as_uint(v));
}

// ---------------------------------------------------------------------------
// f32 -> bf16 streaming convert
// ---------------------------------------------------------------------------
__global__ void k_f32_to_bf16(const float* __restrict__ in,
                              bf16_t* __restrict__ out, int n) {
    int i = blockIdx.x * blockDim.x + threadIdx.x;
    if (i < n) out[i] = f2bf(in[i]);
}

__global__ void k_zero_f32(float* __restrict__ p, int n) {
    int i = blockIdx.x * blockDim.x + threadIdx.x;
    if (i < n) p[i] = 0.f;
}

// ---------------------------------------------------------------------------
// Pack up to 3 row-major [K x colsPer] f32 weight mats into WMMA B-fragment
// order (bf16).  Fragment fi = kc*NtPad + nt holds 32 lanes x 16 elems.
// B 32x16 bf16 layout (ISA 7.12.2): lane l<16 -> col=l, e0..7=K0..7, e8..15=K16..23
//                                   lane l>=16 -> col=l-16, K8..15 / K24..31
// ---------------------------------------------------------------------------
__global__ void k_pack_w(const float* __restrict__ W0,
                         const float* __restrict__ W1,
                         const float* __restrict__ W2,
                         bf16_t* __restrict__ out,
                         int K, int colsPer, int totalCols, int NtPad) {
    int idx = blockIdx.x * blockDim.x + threadIdx.x;
    int total = (K >> 5) * NtPad * 512;
    if (idx >= total) return;
    int e    = idx & 15;
    int lane = (idx >> 4) & 31;
    int fi   = idx >> 9;
    int nt   = fi % NtPad;
    int kc   = fi / NtPad;
    int l15  = lane & 15;
    int hi   = lane >> 4;
    int n = nt * 16 + l15;
    int k = kc * 32 + hi * 8 + e + ((e >> 3) << 3);
    float v = 0.f;
    if (n < totalCols && k < K) {
        int which = n / colsPer;
        int c = n - which * colsPer;
        const float* W = (which == 0) ? W0 : ((which == 1) ? W1 : W2);
        v = W[(size_t)k * colsPer + c];
    }
    out[idx] = f2bf(v);
}

// ---------------------------------------------------------------------------
// WMMA GEMM: C[M x Ncols] = A[M x K](bf16) * Wpacked + bias?
// One wave computes a 16x64 tile (4 accumulators share one A fragment).
// Block = 8 waves = 8 consecutive M-tiles. grid.y covers N in 64-col groups.
// A 16x32 bf16 fragment (ISA 7.12.2): lane<16 row=l, e0..7=K0..7, e8..15=K16..23
//                                     lane>=16 row=l-16, K8..15 / K24..31
// ---------------------------------------------------------------------------
__global__ void k_gemm_wmma(const bf16_t* __restrict__ A,
                            const bf16_t* __restrict__ Wp,
                            float* __restrict__ C,
                            const float* __restrict__ bias,
                            int M, int K, int Ncols, int NtPad, int ldc) {
    const int lane  = threadIdx.x & 31;
    const int wid   = threadIdx.x >> 5;
    const int mtile = blockIdx.x * 8 + wid;
    if (mtile * 16 >= M) return;
    const int ntBase = blockIdx.y * 4;
    const int l15 = lane & 15;
    const int hi  = lane >> 4;

    v8f acc[4];
    acc[0] = v8f{}; acc[1] = v8f{}; acc[2] = v8f{}; acc[3] = v8f{};

    const int row = mtile * 16 + l15;
    const bf16_t* __restrict__ arow = A + (size_t)row * K + hi * 8;
    const int kchunks = K >> 5;

    for (int kc = 0; kc < kchunks; ++kc) {
        v8bf alo = *(const v8bf*)(arow + kc * 32);
        v8bf ahi = *(const v8bf*)(arow + kc * 32 + 16);
        v16bf af;
#pragma unroll
        for (int i = 0; i < 8; ++i) { af[i] = alo[i]; af[i + 8] = ahi[i]; }

        const bf16_t* __restrict__ wb =
            Wp + ((size_t)(kc * NtPad + ntBase) * 32 + lane) * 16;
        v16bf b0 = *(const v16bf*)(wb);
        v16bf b1 = *(const v16bf*)(wb + 32 * 16);
        v16bf b2 = *(const v16bf*)(wb + 2 * 32 * 16);
        v16bf b3 = *(const v16bf*)(wb + 3 * 32 * 16);

        acc[0] = __builtin_amdgcn_wmma_f32_16x16x32_bf16(false, af, false, b0,
                                                         (short)0, acc[0], false, false);
        acc[1] = __builtin_amdgcn_wmma_f32_16x16x32_bf16(false, af, false, b1,
                                                         (short)0, acc[1], false, false);
        acc[2] = __builtin_amdgcn_wmma_f32_16x16x32_bf16(false, af, false, b2,
                                                         (short)0, acc[2], false, false);
        acc[3] = __builtin_amdgcn_wmma_f32_16x16x32_bf16(false, af, false, b3,
                                                         (short)0, acc[3], false, false);
    }

    // C/D layout: lane<16 -> col=l, rows m0+0..7 ; lane>=16 -> col=l-16, rows m0+8..15
    const int rowBase = mtile * 16 + (hi ? 8 : 0);
#pragma unroll
    for (int nt = 0; nt < 4; ++nt) {
        int col = (ntBase + nt) * 16 + l15;
        if (col >= Ncols) continue;
        float bv = bias ? bias[col] : 0.f;
#pragma unroll
        for (int r = 0; r < 8; ++r) {
            int rr = rowBase + r;
            if (rr < M) C[(size_t)rr * ldc + col] = acc[nt][r] + bv;
        }
    }
}

// ---------------------------------------------------------------------------
// H[:,512:768] += (bg + lb)   (constant GAT bias + linear bias, pre-added
// so edge aggregation can accumulate in place)
// ---------------------------------------------------------------------------
__global__ void k_add_bias(float* __restrict__ H, const float* __restrict__ b1,
                           const float* __restrict__ b2, int N) {
    int i = blockIdx.x * blockDim.x + threadIdx.x;
    if (i >= N * HC) return;
    int n = i >> 8, c = i & 255;
    H[(size_t)n * BIGC + 512 + c] += b1[c] + b2[c];
}

// ---------------------------------------------------------------------------
// Per-node attention logits: a_s[n,h]=<xs[n,h,:],atts[h,:]>, same for dst.
// One wave per node, shuffle reduction (wave32). Also inits max/den buffers.
// ---------------------------------------------------------------------------
__global__ void k_attn_node(const float* __restrict__ H,
                            const float* __restrict__ atts,
                            const float* __restrict__ attd,
                            float* __restrict__ a_s, float* __restrict__ a_d,
                            float* __restrict__ mx, float* __restrict__ den,
                            int N) {
    int lane = threadIdx.x & 31;
    int wid  = threadIdx.x >> 5;
    int n = blockIdx.x * 8 + wid;
    if (n >= N) return;
    const float* hrow = H + (size_t)n * BIGC;
#pragma unroll
    for (int h = 0; h < HEADS; ++h) {
        float ps = hrow[h * 64 + lane]        * atts[h * 64 + lane]
                 + hrow[h * 64 + 32 + lane]   * atts[h * 64 + 32 + lane];
        float pd = hrow[256 + h * 64 + lane]      * attd[h * 64 + lane]
                 + hrow[256 + h * 64 + 32 + lane] * attd[h * 64 + 32 + lane];
#pragma unroll
        for (int off = 16; off; off >>= 1) {
            ps += __shfl_xor(ps, off, 32);
            pd += __shfl_xor(pd, off, 32);
        }
        if (lane == 0) { a_s[n * 4 + h] = ps; a_d[n * 4 + h] = pd; }
    }
    if (lane < 4) { mx[n * 4 + lane] = -3.402823466e38f; den[n * 4 + lane] = 0.f; }
}

// ---------------------------------------------------------------------------
// segment-max of leaky_relu(a_s[src]+a_d[dst]) over dst
// ---------------------------------------------------------------------------
__global__ void k_edge_max(const long long* __restrict__ ei,
                           const float* __restrict__ a_s,
                           const float* __restrict__ a_d,
                           float* __restrict__ mx, int E) {
    int idx = blockIdx.x * blockDim.x + threadIdx.x;
    if (idx >= E * HEADS) return;
    int e = idx >> 2, h = idx & 3;
    int s = (int)ei[e], d = (int)ei[E + e];
    float v = a_s[s * 4 + h] + a_d[d * 4 + h];
    v = (v > 0.f) ? v : 0.2f * v;
    atomicMaxF(&mx[d * 4 + h], v);
}

// ---------------------------------------------------------------------------
// ex = exp(e - max[dst]); stash per-edge ex; segment-sum into den[dst]
// ---------------------------------------------------------------------------
__global__ void k_edge_exp(const long long* __restrict__ ei,
                           const float* __restrict__ a_s,
                           const float* __restrict__ a_d,
                           const float* __restrict__ mx,
                           float* __restrict__ den,
                           float* __restrict__ w_e, int E) {
    int idx = blockIdx.x * blockDim.x + threadIdx.x;
    if (idx >= E * HEADS) return;
    int e = idx >> 2, h = idx & 3;
    int s = (int)ei[e], d = (int)ei[E + e];
    float v = a_s[s * 4 + h] + a_d[d * 4 + h];
    v = (v > 0.f) ? v : 0.2f * v;
    float ex = __expf(v - mx[d * 4 + h]);
    w_e[idx] = ex;
    atomicAdd(&den[d * 4 + h], ex);
}

// ---------------------------------------------------------------------------
// Weighted scatter-add of messages: H[dst,512+ch] += xs[src,ch]*w, one wave
// per edge, 8 coalesced 32-lane sweeps over the 256 channels.
// ---------------------------------------------------------------------------
__global__ void k_edge_agg(const long long* __restrict__ ei,
                           const float* __restrict__ w_e,
                           const float* __restrict__ den,
                           float* __restrict__ H, int E) {
    int lane = threadIdx.x & 31;
    int wid  = threadIdx.x >> 5;
    int e = blockIdx.x * 8 + wid;
    if (e >= E) return;
    int s = (int)ei[e], d = (int)ei[E + e];
    const float* srow = H + (size_t)s * BIGC;
    float*       drow = H + (size_t)d * BIGC + 512;
#pragma unroll
    for (int it = 0; it < 8; ++it) {
        int ch = it * 32 + lane;
        int h = ch >> 6;
        float w = w_e[e * 4 + h] / (den[d * 4 + h] + 1e-16f);
        atomicAdd(&drow[ch], srow[ch] * w);
    }
}

// ---------------------------------------------------------------------------
// BatchNorm statistics over nodes (channel-major partials + atomics)
// ---------------------------------------------------------------------------
__global__ void k_bn_stats(const float* __restrict__ H, float* __restrict__ sums,
                           int N) {
    int c = threadIdx.x;                 // 256 channels
    float s = 0.f, s2 = 0.f;
    for (int r = blockIdx.x; r < N; r += gridDim.x) {
        float v = H[(size_t)r * BIGC + 512 + c];
        s += v; s2 += v * v;
    }
    atomicAdd(&sums[c], s);
    atomicAdd(&sums[HC + c], s2);
}

__global__ void k_bn_finalize(const float* __restrict__ sums,
                              const float* __restrict__ g,
                              const float* __restrict__ b,
                              float* __restrict__ scsh, int N) {
    int c = threadIdx.x;
    float inv = 1.f / (float)N;
    float mu  = sums[c] * inv;
    float var = sums[HC + c] * inv - mu * mu;
    float sc  = g[c] * rsqrtf(var + 1e-5f);
    scsh[c]      = sc;
    scsh[HC + c] = b[c] - mu * sc;
}

// BN apply + ReLU + convert to bf16 activations for the next GEMM
__global__ void k_bn_relu_bf16(const float* __restrict__ H,
                               const float* __restrict__ scsh,
                               bf16_t* __restrict__ act, int N) {
    int i = blockIdx.x * blockDim.x + threadIdx.x;
    if (i >= N * HC) return;
    int n = i >> 8, c = i & 255;
    float v = H[(size_t)n * BIGC + 512 + c] * scsh[c] + scsh[HC + c];
    act[i] = f2bf(v > 0.f ? v : 0.f);
}

// ---------------------------------------------------------------------------
// Host side
// ---------------------------------------------------------------------------
extern "C" void kernel_launch(void* const* d_in, const int* in_sizes, int n_in,
                              void* d_out, int out_size, void* d_ws, size_t ws_size,
                              hipStream_t stream) {
    (void)n_in; (void)out_size; (void)ws_size;

    const float*     x  = (const float*)d_in[0];
    const long long* ei = (const long long*)d_in[1];
    const int N = in_sizes[0] / 128;     // 50000
    const int E = in_sizes[1] / 2;       // 800000
    const int K0 = 128;

    const float* ws0   = (const float*)d_in[2];
    const float* wd0   = (const float*)d_in[3];
    const float* atts0 = (const float*)d_in[4];
    const float* attd0 = (const float*)d_in[5];
    const float* bg0   = (const float*)d_in[6];
    const float* lw0   = (const float*)d_in[7];
    const float* lb0   = (const float*)d_in[8];
    const float* g0    = (const float*)d_in[9];
    const float* b0    = (const float*)d_in[10];
    const float* ws1   = (const float*)d_in[11];
    const float* wd1   = (const float*)d_in[12];
    const float* atts1 = (const float*)d_in[13];
    const float* attd1 = (const float*)d_in[14];
    const float* bg1   = (const float*)d_in[15];
    const float* lw1   = (const float*)d_in[16];
    const float* lb1   = (const float*)d_in[17];
    const float* g1    = (const float*)d_in[18];
    const float* b1    = (const float*)d_in[19];
    const float* fcw   = (const float*)d_in[20];
    const float* fcb   = (const float*)d_in[21];

    // --- workspace carve-out (256B aligned) ---
    char* base = (char*)d_ws;
    size_t off = 0;
    auto carve = [&](size_t bytes) -> void* {
        off = (off + 255) & ~(size_t)255;
        void* p = base + off;
        off += bytes;
        return p;
    };
    bf16_t* Xb   = (bf16_t*)carve((size_t)N * K0 * 2);      // bf16 input feats
    bf16_t* ActB = (bf16_t*)carve((size_t)N * HC * 2);      // bf16 hidden acts
    float*  Hbig = (float*) carve((size_t)N * BIGC * 4);    // [xs|xd|pre] fused
    bf16_t* Wp0  = (bf16_t*)carve((size_t)4 * 48 * 512 * 2);
    bf16_t* Wp1  = (bf16_t*)carve((size_t)8 * 48 * 512 * 2);
    bf16_t* WpF  = (bf16_t*)carve((size_t)8 * 24 * 512 * 2);
    float*  a_s  = (float*) carve((size_t)N * 4 * 4);
    float*  a_d  = (float*) carve((size_t)N * 4 * 4);
    float*  mx   = (float*) carve((size_t)N * 4 * 4);
    float*  den  = (float*) carve((size_t)N * 4 * 4);
    float*  w_e  = (float*) carve((size_t)E * 4 * 4);
    float*  sums = (float*) carve(512 * 4);
    float*  scsh = (float*) carve(512 * 4);

    const int Mtiles = (N + 15) / 16;                 // 3125
    const dim3 blk(256);
    const dim3 gemmGrid((Mtiles + 7) / 8, 12);        // 768 cols / 64
    const dim3 gemmGridFC((Mtiles + 7) / 8, 6);       // 384 padded cols / 64

    // --- prep: convert x, pack all weights ---
    k_f32_to_bf16<<<(N * K0 + 255) / 256, blk, 0, stream>>>(x, Xb, N * K0);
    k_pack_w<<<(4 * 48 * 512 + 255) / 256, blk, 0, stream>>>(
        ws0, wd0, lw0, Wp0, K0, HC, BIGC, 48);
    k_pack_w<<<(8 * 48 * 512 + 255) / 256, blk, 0, stream>>>(
        ws1, wd1, lw1, Wp1, HC, HC, BIGC, 48);
    k_pack_w<<<(8 * 24 * 512 + 255) / 256, blk, 0, stream>>>(
        fcw, nullptr, nullptr, WpF, HC, NCLS, NCLS, 24);

    // --- one GAT layer ---
    auto layer = [&](const bf16_t* A, int K, const bf16_t* Wp,
                     const float* atts, const float* attd,
                     const float* bg, const float* lb,
                     const float* g, const float* b) {
        k_gemm_wmma<<<gemmGrid, blk, 0, stream>>>(A, Wp, Hbig, nullptr,
                                                  N, K, BIGC, 48, BIGC);
        k_add_bias<<<(N * HC + 255) / 256, blk, 0, stream>>>(Hbig, bg, lb, N);
        k_attn_node<<<(N + 7) / 8, blk, 0, stream>>>(Hbig, atts, attd,
                                                     a_s, a_d, mx, den, N);
        k_edge_max<<<(E * 4 + 255) / 256, blk, 0, stream>>>(ei, a_s, a_d, mx, E);
        k_edge_exp<<<(E * 4 + 255) / 256, blk, 0, stream>>>(ei, a_s, a_d, mx,
                                                            den, w_e, E);
        k_edge_agg<<<(E + 7) / 8, blk, 0, stream>>>(ei, w_e, den, Hbig, E);
        k_zero_f32<<<2, blk, 0, stream>>>(sums, 512);
        k_bn_stats<<<256, blk, 0, stream>>>(Hbig, sums, N);
        k_bn_finalize<<<1, blk, 0, stream>>>(sums, g, b, scsh, N);
        k_bn_relu_bf16<<<(N * HC + 255) / 256, blk, 0, stream>>>(Hbig, scsh,
                                                                 ActB, N);
    };

    layer(Xb,   K0, Wp0, atts0, attd0, bg0, lb0, g0, b0);
    layer(ActB, HC, Wp1, atts1, attd1, bg1, lb1, g1, b1);

    // --- final FC -> d_out (f32, [N x 349]) ---
    k_gemm_wmma<<<gemmGridFC, blk, 0, stream>>>(ActB, WpF, (float*)d_out, fcb,
                                                N, HC, NCLS, 24, NCLS);
}
